// ViTAttention_9363028705549
// MI455X (gfx1250) — matrix operand order repaired and checked by reading, not compiled, not run on hardware
//
#include <hip/hip_runtime.h>

// ---------------------------------------------------------------------------
// ViT attention for MI455X (gfx1250), wave32.
// - All matrix math via V_WMMA_F32_16X16X32_BF16 (fp32 accumulate).
// - Tile staging via GLOBAL_LOAD_ASYNC_TO_LDS_B128 (ASYNCcnt) + double-
//   buffered LDS so global latency overlaps WMMA compute.
// - Flash-attention structure: scores never touch HBM.
// ---------------------------------------------------------------------------

typedef __bf16 bf16;
typedef __attribute__((ext_vector_type(16))) __bf16 v16bf;
typedef __attribute__((ext_vector_type(8)))  __bf16 v8bf;
typedef __attribute__((ext_vector_type(4)))  __bf16 v4bf;
typedef __attribute__((ext_vector_type(8)))  float  v8f;
typedef __attribute__((ext_vector_type(4)))  float  v4f;

#define Bn   16
#define Nn   1024
#define Cn   768
#define Hn   12
#define Dn   64
#define Mn   (Bn*Nn)     // 16384 tokens
#define O3n  (3*Cn)      // 2304
#define SCALE_F 0.125f   // 64^-0.5

static __device__ __forceinline__ v8f wmma_bf16f32(v16bf a, v16bf b, v8f c) {
  return __builtin_amdgcn_wmma_f32_16x16x32_bf16(false, a, false, b, (short)0, c,
                                                 false, false);
}

// Async global->LDS copy, 16B per lane. VGLOBAL async encoding: VDST = LDS
// byte address (low 32 bits of the generic pointer; flat->LDS truncates to
// addr[31:0]), VADDR = 64-bit global address. Tracked with ASYNCcnt.
static __device__ __forceinline__ void async_b128(const void* g, void* l) {
  unsigned lds = (unsigned)(size_t)l;
  asm volatile("global_load_async_to_lds_b128 %0, %1, off"
               :: "v"(lds), "v"(g) : "memory");
}
static __device__ __forceinline__ void wait_async0() {
  asm volatile("s_wait_asynccnt 0x0" ::: "memory");
}

// -------------------------- fp32 -> bf16 cast ------------------------------
__global__ __launch_bounds__(256) void cast_f32_bf16_x4(
    const float* __restrict__ in, bf16* __restrict__ out, int n4) {
  int i = blockIdx.x * 256 + threadIdx.x;
  if (i < n4) {
    v4f v = *(const v4f*)(in + (size_t)i * 4);
    v4bf o;
    o.x = (bf16)v.x; o.y = (bf16)v.y; o.z = (bf16)v.z; o.w = (bf16)v.w;
    *(v4bf*)(out + (size_t)i * 4) = o;
  }
}

// ------------------------------ GEMM ---------------------------------------
// out[M][Odim] = A[M][768] @ W[Odim][768]^T + bias
// macro tile 128(M) x 64(O), 8 waves as 4x2, each wave 32x32 (2x2 wmma tiles)
// Double-buffered LDS, async tile loads overlap WMMA of the previous k-step.
#define TM 128
#define TO 64
#define TK 32
#define LAS 40   // A LDS row stride (elements): 80B, 16B aligned, bank step 20
#define LBS 40

__global__ __launch_bounds__(256) void gemm_bf16(
    const bf16* __restrict__ A, const bf16* __restrict__ W,
    const float* __restrict__ bias, int Odim, int mode,
    bf16* __restrict__ qkv_out, float* __restrict__ out_f32) {
  __shared__ bf16 lA[2][TM * LAS];   // 2 x 10.0 KB
  __shared__ bf16 lB[2][TO * LBS];   // 2 x  5.0 KB

  const int tid  = threadIdx.x;
  const int wave = tid >> 5;
  const int lane = tid & 31;
  const int l16  = lane & 15;
  const int lh   = lane >> 4;

  const int blockM = blockIdx.y * TM;
  const int blockO = blockIdx.x * TO;
  const int wm = (wave & 3) * 32;
  const int wo = (wave >> 2) * 32;

  const int crow = tid >> 2;           // copy row (0..63)
  const int ccol = (tid & 3) * 8;      // copy col (0/8/16/24)

  auto issue_tile = [&](int kk, int buf) {
#pragma unroll
    for (int i = 0; i < 2; ++i) {      // A slab: 128x32, 2 chunks/thread
      int row = crow + i * 64;
      async_b128(A + (size_t)(blockM + row) * Cn + kk + ccol,
                 &lA[buf][row * LAS + ccol]);
    }
    // W slab: 64x32, 1 chunk/thread
    async_b128(W + (size_t)(blockO + crow) * Cn + kk + ccol,
               &lB[buf][crow * LBS + ccol]);
  };

  issue_tile(0, 0);
  wait_async0();
  __syncthreads();

  v8f acc[2][2] = {};
  int cur = 0;
  for (int kk = 0; kk < Cn; kk += TK) {
    const int nxt = cur ^ 1;
    if (kk + TK < Cn) issue_tile(kk + TK, nxt);            // overlap with WMMA
    if (kk + 2 * TK < Cn)                                  // L2 prefetch ahead
      __builtin_prefetch(A + (size_t)(blockM + (tid >> 1)) * Cn + kk + 2 * TK, 0, 3);

    const bf16* cA = lA[cur];
    const bf16* cB = lB[cur];
    v16bf af[2], bfr[2];
#pragma unroll
    for (int i = 0; i < 2; ++i) {      // A frag: row m=l16, K runs {lh*8, 16+lh*8}
      int row = wm + i * 16 + l16;
      union { v16bf v; v8bf h2[2]; } u;
      u.h2[0] = *(const v8bf*)&cA[row * LAS + lh * 8];
      u.h2[1] = *(const v8bf*)&cA[row * LAS + 16 + lh * 8];
      af[i] = u.v;
    }
#pragma unroll
    for (int j = 0; j < 2; ++j) {      // B frag: col n=l16 (= W row), K runs {lh*8, 16+lh*8}
      int orow = wo + j * 16 + l16;
      union { v16bf v; v8bf h2[2]; } u;
      u.h2[0] = *(const v8bf*)&cB[orow * LBS + lh * 8];
      u.h2[1] = *(const v8bf*)&cB[orow * LBS + 16 + lh * 8];
      bfr[j] = u.v;
    }
#pragma unroll
    for (int i = 0; i < 2; ++i)
#pragma unroll
      for (int j = 0; j < 2; ++j)
        acc[i][j] = wmma_bf16f32(af[i], bfr[j], acc[i][j]);

    wait_async0();
    __syncthreads();
    cur = nxt;
  }

  // epilogue: D layout c[r] = D[r + 8*lh][l16]
#pragma unroll
  for (int i = 0; i < 2; ++i)
#pragma unroll
    for (int j = 0; j < 2; ++j) {
      int o = blockO + wo + j * 16 + l16;
      float bv = bias[o];
#pragma unroll
      for (int r = 0; r < 8; ++r) {
        int m = blockM + wm + i * 16 + lh * 8 + r;
        float v = acc[i][j][r] + bv;
        if (mode == 0) {   // scatter into [3][B][H][N][D] bf16, Q pre-scaled
          int sel = o / Cn;
          int cc  = o - sel * Cn;
          int hh  = cc >> 6;
          int d   = cc & 63;
          int bb  = m >> 10;
          int n   = m & 1023;
          if (sel == 0) v *= SCALE_F;
          qkv_out[(size_t)sel * ((size_t)Bn * Hn * Nn * Dn) +
                  (((size_t)(bb * Hn + hh) * Nn + n) * Dn) + d] = (bf16)v;
        } else {           // fp32 result straight to d_out
          out_f32[(size_t)m * Cn + o] = v;
        }
      }
    }
}

// --------------------------- flash attention -------------------------------
// WG: 8 waves, 128 Q rows (16/wave); 8 KV tiles of 128 tokens, double-buffered:
// async K loads + register-staged transposed V for tile t+1 overlap the
// 32 WMMAs + softmax of tile t.
#define BR 128
#define BC 128
#define KS 72    // K LDS token stride (144B: 16B aligned, bank step 36)
#define VS 136   // V LDS (d-major) token stride (272B: aligned, bank step 4)
#define PS 136   // P LDS row stride

__global__ __launch_bounds__(256) void attn_flash(
    const bf16* __restrict__ Qg, const bf16* __restrict__ Kg,
    const bf16* __restrict__ Vg, bf16* __restrict__ Og) {
  __shared__ bf16 lK[2][BC * KS];        // 2 x 18.0 KB
  __shared__ bf16 lV[2][Dn * VS];        // 2 x 17.0 KB (transposed: [d][token])
  __shared__ bf16 lP[8 * 16 * PS];       // 34.0 KB (per-wave P staging)

  const int tid  = threadIdx.x;
  const int wave = tid >> 5;
  const int lane = tid & 31;
  const int l16  = lane & 15;
  const int lh   = lane >> 4;

  const int bh = blockIdx.x;
  const int b  = bh / Hn;
  const int h  = bh - b * Hn;
  const int qbase = blockIdx.y * BR + wave * 16;

  const bf16* Qb = Qg + (size_t)bh * Nn * Dn;
  const bf16* Kb = Kg + (size_t)bh * Nn * Dn;
  const bf16* Vb = Vg + (size_t)bh * Nn * Dn;

  const int crow = tid >> 3;             // copy row base (0..31)
  const int ccol = (tid & 7) * 8;        // copy col (0..56)

  v16bf qf[2];                           // Q frags, loaded once (D=64 = 2 chunks)
#pragma unroll
  for (int kc = 0; kc < 2; ++kc) {
    union { v16bf v; v8bf h2[2]; } u;
    const bf16* qr = Qb + (size_t)(qbase + l16) * Dn + kc * 32 + lh * 8;
    u.h2[0] = *(const v8bf*)qr;
    u.h2[1] = *(const v8bf*)(qr + 16);
    qf[kc] = u.v;
  }

  // ---- prologue: stage KV tile 0 into buffer 0 ----
  {
    v8bf vr[4];
#pragma unroll
    for (int i = 0; i < 4; ++i) {
      int row = crow + i * 32;
      async_b128(Kb + (size_t)row * Dn + ccol, &lK[0][row * KS + ccol]);
      vr[i] = *(const v8bf*)(Vb + (size_t)row * Dn + ccol);
    }
#pragma unroll
    for (int i = 0; i < 4; ++i) {
      int row = crow + i * 32;
#pragma unroll
      for (int e = 0; e < 8; ++e) lV[0][(ccol + e) * VS + row] = vr[i][e];
    }
  }
  wait_async0();
  __syncthreads();

  v8f oacc[4] = {};
  float mrow[8], lrow[8];
#pragma unroll
  for (int r = 0; r < 8; ++r) { mrow[r] = -1.0e30f; lrow[r] = 0.0f; }

  bf16* pw = &lP[wave * 16 * PS];

  int cur = 0;
  for (int kt = 0; kt < Nn; kt += BC) {
    const int  nxt  = cur ^ 1;
    const bool more = (kt + BC < Nn);

    v8bf vr[4];
    if (more) {                          // stage tile t+1 while computing t
#pragma unroll
      for (int i = 0; i < 4; ++i) {
        int row = crow + i * 32;
        async_b128(Kb + (size_t)(kt + BC + row) * Dn + ccol,
                   &lK[nxt][row * KS + ccol]);
        vr[i] = *(const v8bf*)(Vb + (size_t)(kt + BC + row) * Dn + ccol);
      }
    }

    const bf16* cK = lK[cur];
    const bf16* cV = lV[cur];

    v8f s[8];                            // S row-block: 16 x 128
#pragma unroll
    for (int jc = 0; jc < 8; ++jc) {
      v8f sc = {};
#pragma unroll
      for (int kc = 0; kc < 2; ++kc) {   // B frag = K^T col (= K row, contiguous)
        int tok = jc * 16 + l16;
        union { v16bf v; v8bf h2[2]; } u;
        u.h2[0] = *(const v8bf*)&cK[tok * KS + kc * 32 + lh * 8];
        u.h2[1] = *(const v8bf*)&cK[tok * KS + kc * 32 + lh * 8 + 16];
        sc = wmma_bf16f32(qf[kc], u.v, sc);
      }
      s[jc] = sc;
    }

    // online softmax: each 16-lane half holds rows {r+8*lh}; reduce across n
#pragma unroll
    for (int r = 0; r < 8; ++r) {
      float mx = s[0][r];
#pragma unroll
      for (int jc = 1; jc < 8; ++jc) mx = fmaxf(mx, s[jc][r]);
#pragma unroll
      for (int off = 1; off < 16; off <<= 1)
        mx = fmaxf(mx, __shfl_xor(mx, off, 32));
      float mnew  = fmaxf(mrow[r], mx);
      float alpha = __expf(mrow[r] - mnew);
      mrow[r] = mnew;
      float sum = 0.0f;
#pragma unroll
      for (int jc = 0; jc < 8; ++jc) {
        float p = __expf(s[jc][r] - mnew);
        s[jc][r] = p;
        sum += p;
      }
#pragma unroll
      for (int off = 1; off < 16; off <<= 1)
        sum += __shfl_xor(sum, off, 32);
      lrow[r] = lrow[r] * alpha + sum;
#pragma unroll
      for (int jd = 0; jd < 4; ++jd) oacc[jd][r] *= alpha;
    }

    // D-layout -> A-layout for P via per-wave LDS region (wave-local ordering)
#pragma unroll
    for (int jc = 0; jc < 8; ++jc)
#pragma unroll
      for (int r = 0; r < 8; ++r)
        pw[(r + 8 * lh) * PS + jc * 16 + l16] = (bf16)s[jc][r];

#pragma unroll
    for (int kc = 0; kc < 4; ++kc) {     // O += P (16x128) @ V (128x64)
      union { v16bf v; v8bf h2[2]; } pu;
      pu.h2[0] = *(const v8bf*)&pw[l16 * PS + kc * 32 + lh * 8];
      pu.h2[1] = *(const v8bf*)&pw[l16 * PS + kc * 32 + lh * 8 + 16];
#pragma unroll
      for (int jd = 0; jd < 4; ++jd) {
        union { v16bf v; v8bf h2[2]; } vu;  // B frag = V column d (contig in lV)
        vu.h2[0] = *(const v8bf*)&cV[(jd * 16 + l16) * VS + kc * 32 + lh * 8];
        vu.h2[1] = *(const v8bf*)&cV[(jd * 16 + l16) * VS + kc * 32 + lh * 8 + 16];
        oacc[jd] = wmma_bf16f32(pu.v, vu.v, oacc[jd]);
      }
    }

    if (more) {                          // commit transposed V for tile t+1
#pragma unroll
      for (int i = 0; i < 4; ++i) {
        int row = crow + i * 32;
#pragma unroll
        for (int e = 0; e < 8; ++e) lV[nxt][(ccol + e) * VS + row] = vr[i][e];
      }
    }
    wait_async0();
    __syncthreads();
    cur = nxt;
  }

  // normalize and store as [B*N][768] bf16 (col = h*64+d) for the proj GEMM
#pragma unroll
  for (int jd = 0; jd < 4; ++jd)
#pragma unroll
    for (int r = 0; r < 8; ++r) {
      int n = qbase + 8 * lh + r;
      int d = jd * 16 + l16;
      float v = oacc[jd][r] / lrow[r];
      Og[((size_t)b * Nn + n) * Cn + h * Dn + d] = (bf16)v;
    }
}

// ------------------------------ launch -------------------------------------
extern "C" void kernel_launch(void* const* d_in, const int* in_sizes, int n_in,
                              void* d_out, int out_size, void* d_ws, size_t ws_size,
                              hipStream_t stream) {
  const float* x      = (const float*)d_in[0];
  const float* qkv_w  = (const float*)d_in[1];
  const float* qkv_b  = (const float*)d_in[2];
  const float* proj_w = (const float*)d_in[3];
  const float* proj_b = (const float*)d_in[4];
  float* out = (float*)d_out;
  (void)in_sizes; (void)n_in; (void)out_size; (void)ws_size;

  // workspace layout (~131 MB total)
  char* ws = (char*)d_ws;
  size_t off = 0;
  bf16* x_bf    = (bf16*)(ws + off); off += (size_t)Mn * Cn * sizeof(bf16);
  bf16* wq_bf   = (bf16*)(ws + off); off += (size_t)O3n * Cn * sizeof(bf16);
  bf16* wp_bf   = (bf16*)(ws + off); off += (size_t)Cn * Cn * sizeof(bf16);
  bf16* qkv_bf  = (bf16*)(ws + off); off += (size_t)3 * Bn * Hn * Nn * Dn * sizeof(bf16);
  bf16* attn_bf = (bf16*)(ws + off); off += (size_t)Mn * Cn * sizeof(bf16);

  int nx4 = Mn * Cn / 4;
  cast_f32_bf16_x4<<<(nx4 + 255) / 256, 256, 0, stream>>>(x, x_bf, nx4);
  int nq4 = O3n * Cn / 4;
  cast_f32_bf16_x4<<<(nq4 + 255) / 256, 256, 0, stream>>>(qkv_w, wq_bf, nq4);
  int np4 = Cn * Cn / 4;
  cast_f32_bf16_x4<<<(np4 + 255) / 256, 256, 0, stream>>>(proj_w, wp_bf, np4);

  // QKV: 16384x2304 GEMM, scatter into [3][B][H][N][D] with bias (+ Q*scale)
  gemm_bf16<<<dim3(O3n / TO, Mn / TM), 256, 0, stream>>>(
      x_bf, wq_bf, qkv_b, O3n, 0, qkv_bf, nullptr);

  size_t BHND = (size_t)Bn * Hn * Nn * Dn;
  attn_flash<<<dim3(Bn * Hn, Nn / BR), 256, 0, stream>>>(
      qkv_bf, qkv_bf + BHND, qkv_bf + 2 * BHND, attn_bf);

  // proj: 16384x768 GEMM, fp32 output straight into d_out
  gemm_bf16<<<dim3(Cn / TO, Mn / TM), 256, 0, stream>>>(
      attn_bf, wp_bf, proj_b, Cn, 1, nullptr, out);
}
